// Mamba_55198919688487
// MI455X (gfx1250) — matrix operand order repaired and checked by reading, not compiled, not run on hardware
//
#include <hip/hip_runtime.h>
#include <hip/hip_bf16.h>

// ---------------------------------------------------------------------------
// Mamba block forward for MI455X (gfx1250), wave32 + WMMA + TDM.
//
//  1) convert x / weights to f16 in ws
//  2) WMMA GEMM (TDM-staged B panel in LDS):  xz = x @ in_proj_w.T
//  3) causal depthwise conv(4) + silu -> xi (f32 + f16)
//  4) WMMA GEMM (simple):  x_dbl = xi @ x_proj_w.T   (N=96)
//  5) WMMA GEMM (TDM):     dt_raw = x_dbl[:, :64] @ dt_proj_w.T
//  6) delta = softplus(dt_raw + dt_proj_b)  (in place)
//  7) chunked selective scan, 3 passes (16 chunks of 64 steps)
//  8) WMMA GEMM (TDM):     out = yh @ out_proj_w.T -> d_out f32
// ---------------------------------------------------------------------------

#define D_MODEL 1024
#define D_STATE 16
#define D_CONV 4
#define D_INNER 2048
#define DT_RANK 64
#define SEQ_L 1024
#define CHUNKS 16
#define CLEN (SEQ_L / CHUNKS)  // 64

typedef _Float16 v16h __attribute__((ext_vector_type(16)));
typedef _Float16 v8h  __attribute__((ext_vector_type(8)));
typedef float    v8f  __attribute__((ext_vector_type(8)));

#if defined(__has_builtin)
#  if __has_builtin(__builtin_amdgcn_tensor_load_to_lds)
#    define HAVE_TDM 1
#  endif
#endif
#if defined(__has_include)
#  if __has_include(<hip/amd_detail/amd_gfx1250_TDM.h>)
#    define TDM_6ARG 1
#  endif
#endif

// ---------------------------------------------------------------- converts --
__global__ void f32_to_f16_kernel(const float* __restrict__ s,
                                  _Float16* __restrict__ d, int n) {
  for (int i = blockIdx.x * blockDim.x + threadIdx.x; i < n;
       i += gridDim.x * blockDim.x)
    d[i] = (_Float16)s[i];
}

__global__ void rows_f32_to_f16_kernel(const float* __restrict__ s, int sld,
                                       _Float16* __restrict__ d, int dld,
                                       int rows, int cols) {
  int i = blockIdx.x * blockDim.x + threadIdx.x;
  if (i >= rows * cols) return;
  int r = i / cols, c = i - r * cols;
  d[(size_t)r * dld + c] = (_Float16)s[(size_t)r * sld + c];
}

__global__ void aneg_kernel(const float* __restrict__ a_log,
                            float* __restrict__ an, int n) {
  int i = blockIdx.x * blockDim.x + threadIdx.x;
  if (i < n) an[i] = -__expf(a_log[i]);
}

// -------------------------------------------------- WMMA fragment helpers --
__device__ __forceinline__ v16h combine8(v8h lo, v8h hi) {
  return __builtin_shufflevector(lo, hi, 0, 1, 2, 3, 4, 5, 6, 7,
                                 8, 9, 10, 11, 12, 13, 14, 15);
}
__device__ __forceinline__ v16h load_a_frag(const _Float16* row, int k0, int half) {
  v8h alo = *(const v8h*)(row + k0 + 16 * half);
  v8h ahi = *(const v8h*)(row + k0 + 16 * half + 8);
  return combine8(alo, ahi);
}

// ------------------------------------------ WMMA GEMM, simple (1 tile/wave) -
// C[M,N] = A[M,K] @ W[N,K]^T.  Used only for the N=96 x_proj GEMM.
__global__ void gemm_f16_wmma_kernel(const _Float16* __restrict__ A,
                                     const _Float16* __restrict__ W,
                                     float* __restrict__ C,
                                     int M, int N, int K) {
  const int wave = (blockIdx.x * blockDim.x + threadIdx.x) >> 5;
  const int lane = threadIdx.x & 31;
  const int ntiles = N >> 4;
  const int mt = wave / ntiles;
  const int nt = wave - mt * ntiles;
  if (mt >= (M >> 4)) return;  // wave-uniform (EXEC stays all-ones)

  const int mn = lane & 15, half = lane >> 4;
  const _Float16* arow = A + (size_t)(mt * 16 + mn) * K;
  const _Float16* wrow = W + (size_t)(nt * 16 + mn) * K;

  v8f acc = {};
  for (int k0 = 0; k0 < K; k0 += 32) {
    v16h a = load_a_frag(arow, k0, half);
    v16h b = *(const v16h*)(wrow + k0 + 16 * half);
    __builtin_prefetch(arow + k0 + 64, 0, 3);
    __builtin_prefetch(wrow + k0 + 64, 0, 3);
    acc = __builtin_amdgcn_wmma_f32_16x16x32_f16(false, a, false, b,
                                                 (short)0, acc, false, false);
  }
  float* crow = C + (size_t)(mt * 16 + 8 * half) * N + nt * 16 + mn;
#pragma unroll
  for (int r = 0; r < 8; ++r) crow[(size_t)r * N] = acc[r];
}

// ------------------------------- WMMA GEMM, 2(M) x 4(N) tiles per wave -----
// Pure-global fallback (no TDM builtin available).
__global__ void gemm_f16_wmma_2x4_kernel(const _Float16* __restrict__ A,
                                         const _Float16* __restrict__ W,
                                         float* __restrict__ C,
                                         int M, int N, int K) {
  const int wave = (blockIdx.x * blockDim.x + threadIdx.x) >> 5;
  const int lane = threadIdx.x & 31;
  const int ngrp = N >> 6;
  const int mg = wave / ngrp;
  const int ng = wave - mg * ngrp;
  if (mg >= (M >> 5)) return;  // wave-uniform

  const int mn = lane & 15, half = lane >> 4;
  const _Float16* arow0 = A + (size_t)(mg * 32 + mn) * K;
  const _Float16* arow1 = arow0 + (size_t)16 * K;
  const _Float16* wbase = W + (size_t)(ng * 64 + mn) * K;

  v8f acc[2][4];
#pragma unroll
  for (int mt = 0; mt < 2; ++mt)
#pragma unroll
    for (int nt = 0; nt < 4; ++nt) acc[mt][nt] = {};

  for (int k0 = 0; k0 < K; k0 += 32) {
    v16h a0 = load_a_frag(arow0, k0, half);
    v16h a1 = load_a_frag(arow1, k0, half);
    v16h b[4];
#pragma unroll
    for (int nt = 0; nt < 4; ++nt)
      b[nt] = *(const v16h*)(wbase + (size_t)nt * 16 * K + k0 + 16 * half);
    __builtin_prefetch(arow0 + k0 + 64, 0, 3);
    __builtin_prefetch(arow1 + k0 + 64, 0, 3);
#pragma unroll
    for (int nt = 0; nt < 4; ++nt) {
      acc[0][nt] = __builtin_amdgcn_wmma_f32_16x16x32_f16(
          false, a0, false, b[nt], (short)0, acc[0][nt], false, false);
      acc[1][nt] = __builtin_amdgcn_wmma_f32_16x16x32_f16(
          false, a1, false, b[nt], (short)0, acc[1][nt], false, false);
    }
  }
#pragma unroll
  for (int mt = 0; mt < 2; ++mt)
#pragma unroll
    for (int nt = 0; nt < 4; ++nt) {
      float* crow = C + (size_t)(mg * 32 + mt * 16 + 8 * half) * N +
                    ng * 64 + nt * 16 + mn;
#pragma unroll
      for (int r = 0; r < 8; ++r) crow[(size_t)r * N] = acc[mt][nt][r];
    }
}

#ifdef HAVE_TDM
// ----------------------------- WMMA GEMM with TDM-staged B panel in LDS ----
// Block (8 waves) computes 256(M) x 64(N); the 64 x Kc B panel is DMA'd to
// LDS once per k-chunk by the Tensor Data Mover (double-buffered, TENSORcnt),
// with a 4-DWORD row pad so 16-row ds reads are bank-conflict free.
typedef unsigned int tdm_v4u __attribute__((ext_vector_type(4)));
typedef int          tdm_v4i __attribute__((ext_vector_type(4)));
typedef int          tdm_v8i __attribute__((ext_vector_type(8)));

__device__ __forceinline__ void tdm_load_b_tile(const _Float16* gsrc,
                                                unsigned lds_off,
                                                int K, int N, int Kc) {
  unsigned long long ga = (unsigned long long)(uintptr_t)gsrc;
  tdm_v4u g0;
  g0.x = 1u;                                     // count=1 (valid user D#)
  g0.y = lds_off;                                // lds_addr
  g0.z = (unsigned)ga;                           // global_addr[31:0]
  g0.w = ((unsigned)(ga >> 32) & 0x01FFFFFFu) | 0x80000000u;  // [56:32]+type=2
  const int padInt = (Kc == 128) ? 5 : 4;        // 64 / 32 DWORDs per row
  tdm_v8i g1;
  g1[0] = (1 << 16) | (1 << 20) | (padInt << 22) | (3 << 25);
  //        data_size=2B  pad_en     pad_interval    pad_amount=4 DWORDs
  g1[1] = (K & 0xFFFF) << 16;                    // tensor_dim0 lo
  g1[2] = ((K >> 16) & 0xFFFF) | ((N & 0xFFFF) << 16);   // dim0 hi, dim1 lo
  g1[3] = ((N >> 16) & 0xFFFF) | (Kc << 16);     // dim1 hi, tile_dim0=Kc
  g1[4] = 64;                                    // tile_dim1=64, tile_dim2=0
  g1[5] = K;                                     // tensor_dim0_stride lo
  g1[6] = 0;                                     // stride hi, dim1_stride lo
  g1[7] = 0;
  tdm_v4i z4 = {0, 0, 0, 0};
#ifdef TDM_6ARG
  tdm_v8i z8 = {};
  __builtin_amdgcn_tensor_load_to_lds(g0, g1, z4, z4, z8, 0);
#else
  __builtin_amdgcn_tensor_load_to_lds(g0, g1, z4, z4, 0);
#endif
}

#define MAX_KC 128
__global__ void gemm_f16_wmma_tdm_kernel(const _Float16* __restrict__ A,
                                         const _Float16* __restrict__ W,
                                         float* __restrict__ C,
                                         int M, int N, int K, int Kc) {
  __shared__ _Float16 Bs[2][64 * (MAX_KC + 8)];
  const int widx = threadIdx.x >> 5;
  const int lane = threadIdx.x & 31;
  const int ngrp = N >> 6;
  const int bm = blockIdx.x / ngrp;        // 256-row block index
  const int ng = blockIdx.x - bm * ngrp;   // 64-col group
  const int mg = bm * 8 + widx;            // this wave's 32-row group
  const int mn = lane & 15, half = lane >> 4;
  const int rowH = Kc + 8;                 // padded halves per LDS row

  const _Float16* arow0 = A + (size_t)(mg * 32 + mn) * K;
  const _Float16* arow1 = arow0 + (size_t)16 * K;
  const _Float16* wtile = W + (size_t)(ng * 64) * K;

  v8f acc[2][4];
#pragma unroll
  for (int mt = 0; mt < 2; ++mt)
#pragma unroll
    for (int nt = 0; nt < 4; ++nt) acc[mt][nt] = {};

  const int nc = K / Kc;
  if (widx == 0)
    tdm_load_b_tile(wtile, (unsigned)(uintptr_t)&Bs[0][0], K, N, Kc);

  for (int c = 0; c < nc; ++c) {
    if (widx == 0) {
      if (c + 1 < nc) {
        tdm_load_b_tile(wtile + (size_t)(c + 1) * Kc,
                        (unsigned)(uintptr_t)&Bs[(c + 1) & 1][0], K, N, Kc);
        __builtin_amdgcn_s_wait_tensorcnt(1);  // chunk c done, c+1 in flight
      } else {
        __builtin_amdgcn_s_wait_tensorcnt(0);
      }
    }
    __syncthreads();  // publish chunk c to all waves
    const _Float16* bbuf = &Bs[c & 1][0];
    const int k0 = c * Kc;
    for (int kl = 0; kl < Kc; kl += 32) {
      v16h a0 = load_a_frag(arow0, k0 + kl, half);
      v16h a1 = load_a_frag(arow1, k0 + kl, half);
      __builtin_prefetch(arow0 + k0 + kl + 64, 0, 3);
      __builtin_prefetch(arow1 + k0 + kl + 64, 0, 3);
      v16h b[4];
#pragma unroll
      for (int nt = 0; nt < 4; ++nt) {
        const _Float16* br = bbuf + (size_t)(nt * 16 + mn) * rowH + kl + 16 * half;
        v8h blo = *(const v8h*)br;        // ds_load_b128
        v8h bhi = *(const v8h*)(br + 8);  // ds_load_b128
        b[nt] = combine8(blo, bhi);
      }
#pragma unroll
      for (int nt = 0; nt < 4; ++nt) {
        acc[0][nt] = __builtin_amdgcn_wmma_f32_16x16x32_f16(
            false, a0, false, b[nt], (short)0, acc[0][nt], false, false);
        acc[1][nt] = __builtin_amdgcn_wmma_f32_16x16x32_f16(
            false, a1, false, b[nt], (short)0, acc[1][nt], false, false);
      }
    }
    __syncthreads();  // all reads of chunk c done before buffer reuse
  }
#pragma unroll
  for (int mt = 0; mt < 2; ++mt)
#pragma unroll
    for (int nt = 0; nt < 4; ++nt) {
      float* crow = C + (size_t)(mg * 32 + mt * 16 + 8 * half) * N +
                    ng * 64 + nt * 16 + mn;
#pragma unroll
      for (int r = 0; r < 8; ++r) crow[(size_t)r * N] = acc[mt][nt][r];
    }
}
#endif  // HAVE_TDM

// ------------------------------------------------------------ conv + silu --
__global__ void conv_silu_kernel(const float* __restrict__ xz,   // [L,4096]
                                 const float* __restrict__ cw,   // [2048,4]
                                 const float* __restrict__ cb,   // [2048]
                                 float* __restrict__ xi,         // [L,2048]
                                 _Float16* __restrict__ xih) {   // [L,2048]
  int idx = blockIdx.x * blockDim.x + threadIdx.x;
  if (idx >= SEQ_L * D_INNER) return;
  int l = idx >> 11;
  int d = idx & (D_INNER - 1);
  float acc = cb[d];
#pragma unroll
  for (int t = 0; t < D_CONV; ++t) {
    int ls = l - (D_CONV - 1) + t;
    if (ls >= 0) acc += cw[d * D_CONV + t] * xz[(size_t)ls * (2 * D_INNER) + d];
  }
  float s = acc / (1.f + __expf(-acc));  // silu
  xi[idx] = s;
  xih[idx] = (_Float16)s;
}

// ---------------------------------------------------------- delta softplus --
__global__ void delta_softplus_kernel(float* __restrict__ dt,
                                      const float* __restrict__ bias) {
  int idx = blockIdx.x * blockDim.x + threadIdx.x;
  if (idx >= SEQ_L * D_INNER) return;
  float v = dt[idx] + bias[idx & (D_INNER - 1)];
  dt[idx] = (v > 20.f) ? v : log1pf(__expf(v));
}

// -------------------------------------------------- chunked selective scan --
__global__ void scan_pass1_kernel(const float* __restrict__ delta,
                                  const float* __restrict__ xi,
                                  const float* __restrict__ xdbl,
                                  const float* __restrict__ Aneg,
                                  float* __restrict__ P,
                                  float* __restrict__ S) {
  const int wave = (blockIdx.x * blockDim.x + threadIdx.x) >> 5;
  const int lane = threadIdx.x & 31;
  const int c = wave / (D_INNER / 32);
  const int db = wave - c * (D_INNER / 32);
  if (c >= CHUNKS) return;
  const int d = db * 32 + lane;

  float Areg[D_STATE], st[D_STATE], pr[D_STATE];
#pragma unroll
  for (int n = 0; n < D_STATE; ++n) {
    Areg[n] = Aneg[(size_t)d * D_STATE + n];
    st[n] = 0.f;
    pr[n] = 1.f;
  }
  const int l0 = c * CLEN;
  for (int l = l0; l < l0 + CLEN; ++l) {
    const float dl = delta[(size_t)l * D_INNER + d];
    const float du = dl * xi[(size_t)l * D_INNER + d];
    const float* bc = xdbl + (size_t)l * (DT_RANK + 2 * D_STATE);
#pragma unroll
    for (int n = 0; n < D_STATE; ++n) {
      float dA = __expf(dl * Areg[n]);  // v_exp_f32
      st[n] = dA * st[n] + du * bc[DT_RANK + n];
      pr[n] *= dA;
    }
  }
  const size_t base = (size_t)c * D_INNER * D_STATE + (size_t)d * D_STATE;
#pragma unroll
  for (int n = 0; n < D_STATE; ++n) {
    P[base + n] = pr[n];
    S[base + n] = st[n];
  }
}

__global__ void scan_pass2_kernel(const float* __restrict__ P,
                                  const float* __restrict__ S,
                                  float* __restrict__ init) {
  const int i = blockIdx.x * blockDim.x + threadIdx.x;
  if (i >= D_INNER * D_STATE) return;
  float run = 0.f;
  for (int c = 0; c < CHUNKS; ++c) {
    const size_t o = (size_t)c * D_INNER * D_STATE + i;
    init[o] = run;
    run = P[o] * run + S[o];
  }
}

__global__ void scan_pass3_kernel(const float* __restrict__ delta,
                                  const float* __restrict__ xi,
                                  const float* __restrict__ xz,    // z part
                                  const float* __restrict__ xdbl,
                                  const float* __restrict__ Aneg,
                                  const float* __restrict__ Dv,
                                  const float* __restrict__ init,
                                  _Float16* __restrict__ yh) {
  const int wave = (blockIdx.x * blockDim.x + threadIdx.x) >> 5;
  const int lane = threadIdx.x & 31;
  const int c = wave / (D_INNER / 32);
  const int db = wave - c * (D_INNER / 32);
  if (c >= CHUNKS) return;
  const int d = db * 32 + lane;

  float Areg[D_STATE], st[D_STATE];
  const size_t base = (size_t)c * D_INNER * D_STATE + (size_t)d * D_STATE;
#pragma unroll
  for (int n = 0; n < D_STATE; ++n) {
    Areg[n] = Aneg[(size_t)d * D_STATE + n];
    st[n] = init[base + n];
  }
  const float Dd = Dv[d];
  const int l0 = c * CLEN;
  for (int l = l0; l < l0 + CLEN; ++l) {
    const float dl = delta[(size_t)l * D_INNER + d];
    const float ul = xi[(size_t)l * D_INNER + d];
    const float zl = xz[(size_t)l * (2 * D_INNER) + D_INNER + d];
    const float du = dl * ul;
    const float* bc = xdbl + (size_t)l * (DT_RANK + 2 * D_STATE);
    float y = 0.f;
#pragma unroll
    for (int n = 0; n < D_STATE; ++n) {
      float dA = __expf(dl * Areg[n]);
      st[n] = dA * st[n] + du * bc[DT_RANK + n];
      y += st[n] * bc[DT_RANK + D_STATE + n];
    }
    y += ul * Dd;
    y *= zl / (1.f + __expf(-zl));  // silu(z)
    yh[(size_t)l * D_INNER + d] = (_Float16)y;
  }
}

// ---------------------------------------------------------------------------
static inline size_t align_up(size_t v, size_t a) { return (v + a - 1) / a * a; }

extern "C" void kernel_launch(void* const* d_in, const int* in_sizes, int n_in,
                              void* d_out, int out_size, void* d_ws, size_t ws_size,
                              hipStream_t stream) {
  const float* x         = (const float*)d_in[0];
  const float* in_proj_w = (const float*)d_in[1];
  const float* conv_w    = (const float*)d_in[2];
  const float* conv_b    = (const float*)d_in[3];
  const float* x_proj_w  = (const float*)d_in[4];
  const float* dt_proj_w = (const float*)d_in[5];
  const float* dt_proj_b = (const float*)d_in[6];
  const float* A_log     = (const float*)d_in[7];
  const float* Dvec      = (const float*)d_in[8];
  const float* out_proj_w= (const float*)d_in[9];
  float* out = (float*)d_out;

  char* ws = (char*)d_ws;
  size_t cur = 0;
  auto alloc = [&](size_t bytes) -> char* {
    char* p = ws + cur;
    cur = align_up(cur + bytes, 256);
    return p;
  };

  _Float16* xh     = (_Float16*)alloc((size_t)SEQ_L * D_MODEL * 2);
  _Float16* w_in_h = (_Float16*)alloc((size_t)2 * D_INNER * D_MODEL * 2);
  _Float16* w_xp_h = (_Float16*)alloc((size_t)(DT_RANK + 2 * D_STATE) * D_INNER * 2);
  _Float16* w_dt_h = (_Float16*)alloc((size_t)D_INNER * DT_RANK * 2);
  _Float16* w_out_h= (_Float16*)alloc((size_t)D_MODEL * D_INNER * 2);
  float*    aneg   = (float*)   alloc((size_t)D_INNER * D_STATE * 4);
  float*    xz     = (float*)   alloc((size_t)SEQ_L * 2 * D_INNER * 4);
  float*    xi     = (float*)   alloc((size_t)SEQ_L * D_INNER * 4);
  _Float16* xi_h   = (_Float16*)alloc((size_t)SEQ_L * D_INNER * 2);
  float*    xdbl   = (float*)   alloc((size_t)SEQ_L * (DT_RANK + 2 * D_STATE) * 4);
  _Float16* dtin_h = (_Float16*)alloc((size_t)SEQ_L * DT_RANK * 2);
  float*    dtout  = (float*)   alloc((size_t)SEQ_L * D_INNER * 4);  // -> delta
  _Float16* y_h    = (_Float16*)alloc((size_t)SEQ_L * D_INNER * 2);
  float*    Pbuf   = (float*)   alloc((size_t)CHUNKS * D_INNER * D_STATE * 4);
  float*    Sbuf   = (float*)   alloc((size_t)CHUNKS * D_INNER * D_STATE * 4);
  float*    ibuf   = (float*)   alloc((size_t)CHUNKS * D_INNER * D_STATE * 4);

  const int TPB = 256;
  f32_to_f16_kernel<<<1024, TPB, 0, stream>>>(x, xh, SEQ_L * D_MODEL);
  f32_to_f16_kernel<<<2048, TPB, 0, stream>>>(in_proj_w, w_in_h, 2 * D_INNER * D_MODEL);
  f32_to_f16_kernel<<<256, TPB, 0, stream>>>(x_proj_w, w_xp_h,
                                             (DT_RANK + 2 * D_STATE) * D_INNER);
  f32_to_f16_kernel<<<256, TPB, 0, stream>>>(dt_proj_w, w_dt_h, D_INNER * DT_RANK);
  f32_to_f16_kernel<<<1024, TPB, 0, stream>>>(out_proj_w, w_out_h, D_MODEL * D_INNER);
  aneg_kernel<<<(D_INNER * D_STATE + TPB - 1) / TPB, TPB, 0, stream>>>(
      A_log, aneg, D_INNER * D_STATE);

  // in_proj: [1024,1024] x [4096,1024]^T -> [1024,4096]
#ifdef HAVE_TDM
  gemm_f16_wmma_tdm_kernel<<<(SEQ_L / 256) * (2 * D_INNER / 64), TPB, 0, stream>>>(
      xh, w_in_h, xz, SEQ_L, 2 * D_INNER, D_MODEL, 128);
#else
  gemm_f16_wmma_2x4_kernel<<<((SEQ_L / 32) * (2 * D_INNER / 64)) / 8, TPB, 0, stream>>>(
      xh, w_in_h, xz, SEQ_L, 2 * D_INNER, D_MODEL);
#endif
  conv_silu_kernel<<<(SEQ_L * D_INNER + TPB - 1) / TPB, TPB, 0, stream>>>(
      xz, conv_w, conv_b, xi, xi_h);
  // x_proj (N=96 -> simple kernel)
  {
    int tiles = (SEQ_L / 16) * ((DT_RANK + 2 * D_STATE) / 16);
    gemm_f16_wmma_kernel<<<(tiles + 7) / 8, TPB, 0, stream>>>(
        xi_h, w_xp_h, xdbl, SEQ_L, DT_RANK + 2 * D_STATE, D_INNER);
  }
  rows_f32_to_f16_kernel<<<(SEQ_L * DT_RANK + TPB - 1) / TPB, TPB, 0, stream>>>(
      xdbl, DT_RANK + 2 * D_STATE, dtin_h, DT_RANK, SEQ_L, DT_RANK);
  // dt_proj: K=64 -> single TDM chunk of Kc=64
#ifdef HAVE_TDM
  gemm_f16_wmma_tdm_kernel<<<(SEQ_L / 256) * (D_INNER / 64), TPB, 0, stream>>>(
      dtin_h, w_dt_h, dtout, SEQ_L, D_INNER, DT_RANK, 64);
#else
  gemm_f16_wmma_2x4_kernel<<<((SEQ_L / 32) * (D_INNER / 64)) / 8, TPB, 0, stream>>>(
      dtin_h, w_dt_h, dtout, SEQ_L, D_INNER, DT_RANK);
#endif
  delta_softplus_kernel<<<(SEQ_L * D_INNER + TPB - 1) / TPB, TPB, 0, stream>>>(
      dtout, dt_proj_b);
  // chunked scan
  {
    int waves = CHUNKS * (D_INNER / 32);
    scan_pass1_kernel<<<waves / 8, TPB, 0, stream>>>(dtout, xi, xdbl, aneg,
                                                     Pbuf, Sbuf);
    scan_pass2_kernel<<<(D_INNER * D_STATE) / TPB, TPB, 0, stream>>>(
        Pbuf, Sbuf, ibuf);
    scan_pass3_kernel<<<waves / 8, TPB, 0, stream>>>(dtout, xi, xz, xdbl, aneg,
                                                     Dvec, ibuf, y_h);
  }
  // out_proj
#ifdef HAVE_TDM
  gemm_f16_wmma_tdm_kernel<<<(SEQ_L / 256) * (D_MODEL / 64), TPB, 0, stream>>>(
      y_h, w_out_h, out, SEQ_L, D_MODEL, D_INNER, 128);
#else
  gemm_f16_wmma_2x4_kernel<<<((SEQ_L / 32) * (D_MODEL / 64)) / 8, TPB, 0, stream>>>(
      y_h, w_out_h, out, SEQ_L, D_MODEL, D_INNER);
#endif
  (void)in_sizes; (void)n_in; (void)out_size; (void)ws_size;
}
